// TMSA_21474836480654
// MI455X (gfx1250) — matrix operand order repaired and checked by reading, not compiled, not run on hardware
//
#include <hip/hip_runtime.h>

typedef unsigned short u16;
typedef __attribute__((ext_vector_type(16))) __bf16 v16bf;
typedef __attribute__((ext_vector_type(8)))  float  v8f;

struct alignas(16) F4 { float x, y, z, w; };

union Frag {
  v16bf v;
  F4    f[2];
};

__device__ __forceinline__ u16 f2bf(float f) {
  unsigned u = __float_as_uint(f);
  u += 0x7FFFu + ((u >> 16) & 1u);   // round-to-nearest-even
  return (u16)(u >> 16);
}

__device__ __forceinline__ v8f vzero() {
  v8f z;
#pragma unroll
  for (int i = 0; i < 8; ++i) z[i] = 0.f;
  return z;
}

__device__ __forceinline__ v8f wmma_bf16(const Frag& a, const Frag& b, v8f c) {
  // D = A(16x32 bf16) * B(32x16 bf16) + C(16x16 f32)
  return __builtin_amdgcn_wmma_f32_16x16x32_bf16(false, a.v, false, b.v,
                                                 (short)0, c, false, false);
}

// ---------------------------------------------------------------------------
// Prep: xs||xt -> Xcat bf16 [8192 x 1024]
// ---------------------------------------------------------------------------
__global__ __launch_bounds__(256)
void convert_x(const float* __restrict__ xs, const float* __restrict__ xt,
               u16* __restrict__ dst) {
  const int idx = blockIdx.x * 256 + threadIdx.x;     // 8192*1024 threads
  const int m = idx >> 10;
  const int k = idx & 1023;
  const float v = (k < 512) ? xs[(size_t)m * 512 + k]
                            : xt[(size_t)m * 512 + (k - 512)];
  dst[idx] = f2bf(v);
}

// Wt[n][k] = (k<512 ? Ws[k][n] : Wt2[k-512][n]) as bf16; N=512, stride Ktot
__global__ __launch_bounds__(256)
void convert_w(const float* __restrict__ s1, const float* __restrict__ s2,
               u16* __restrict__ dst, int Ktot) {
  const int idx = blockIdx.x * 256 + threadIdx.x;     // 512*Ktot threads
  const int n = idx / Ktot;
  const int k = idx - n * Ktot;
  const float v = (k < 512) ? s1[(size_t)k * 512 + n]
                            : s2[(size_t)(k - 512) * 512 + n];
  dst[idx] = f2bf(v);
}

// ---------------------------------------------------------------------------
// GEMM: C[8192 x 512] = X[8192 x Ktot] * W[Ktot x 512]  (W staged as Wt[n][k])
// block tile 128x128, 4 waves, each wave 64x64 (4x4 WMMA tiles), K step 32.
// Software-pipelined: next K-tile register-staged during WMMA, double LDS buf.
// MODE: 0 = bf16 out, 1 = bf16 out + WK bias (K projection), 2 = f32 out.
// ---------------------------------------------------------------------------
#define AS_STR 40   // padded LDS row stride in bf16 elements (80B, 16B-aligned)

template <int MODE>
__global__ __launch_bounds__(128)
void gemm_bf16(const u16* __restrict__ X, const u16* __restrict__ Wt,
               u16* __restrict__ outB, float* __restrict__ outF,
               const float* __restrict__ WKb, int Ktot) {
  __shared__ u16 As[2][128 * AS_STR];
  __shared__ u16 Bs[2][128 * AS_STR];

  const int tid  = threadIdx.x;
  const int lane = tid & 31;
  const int wave = tid >> 5;
  const int m0 = blockIdx.x * 128;
  const int n0 = blockIdx.y * 128;
  const int wm = (wave >> 1) * 64;
  const int wn = (wave & 1) * 64;
  const int l16 = lane & 15;
  const int hi  = lane >> 4;

  v8f acc[4][4];
#pragma unroll
  for (int t = 0; t < 4; ++t)
#pragma unroll
    for (int u = 0; u < 4; ++u) acc[t][u] = vzero();

  // prologue: stage K-tile 0 into registers
  const u16* rowA = X  + (size_t)(m0 + tid) * Ktot;
  const u16* rowB = Wt + (size_t)(n0 + tid) * Ktot;
  F4 ra[4], rb[4];
#pragma unroll
  for (int i = 0; i < 4; ++i) {
    ra[i] = ((const F4*)rowA)[i];
    rb[i] = ((const F4*)rowB)[i];
  }

  const int nk = Ktot >> 5;
  for (int kk = 0; kk < nk; ++kk) {
    const int buf = kk & 1;
    F4* sA = (F4*)(As[buf] + tid * AS_STR);
    F4* sB = (F4*)(Bs[buf] + tid * AS_STR);
#pragma unroll
    for (int i = 0; i < 4; ++i) { sA[i] = ra[i]; sB[i] = rb[i]; }
    __syncthreads();

    // issue next tile's global loads; they overlap with this tile's WMMAs
    if (kk + 1 < nk) {
      const F4* gA = (const F4*)(rowA + (kk + 1) * 32);
      const F4* gB = (const F4*)(rowB + (kk + 1) * 32);
#pragma unroll
      for (int i = 0; i < 4; ++i) { ra[i] = gA[i]; rb[i] = gB[i]; }
    }

    Frag a[4], b[4];
#pragma unroll
    for (int t = 0; t < 4; ++t) {
      // A layout: lane m=l16, k = hi*16 + 0..15
      const u16* p = As[buf] + (wm + t * 16 + l16) * AS_STR;
      a[t].f[0] = *(const F4*)(p + hi * 16);
      a[t].f[1] = *(const F4*)(p + hi * 16 + 8);
    }
#pragma unroll
    for (int u = 0; u < 4; ++u) {
      // B layout: lane n=l16, k = hi*16 + 0..15 (contiguous in Wt[n][k])
      const u16* p = Bs[buf] + (wn + u * 16 + l16) * AS_STR;
      b[u].f[0] = *(const F4*)(p + hi * 16);
      b[u].f[1] = *(const F4*)(p + hi * 16 + 8);
    }
#pragma unroll
    for (int t = 0; t < 4; ++t)
#pragma unroll
      for (int u = 0; u < 4; ++u)
        acc[t][u] = wmma_bf16(a[t], b[u], acc[t][u]);
    // no trailing barrier: double buffer + per-wave in-order LDS +
    // pre-barrier dscnt wait protect the buffer being refilled next iter
  }

  // epilogue (straight-line, MODE folded at compile time)
  // C layout: lane n=l16, vgpr r -> m = r + hi*8
#pragma unroll
  for (int t = 0; t < 4; ++t)
#pragma unroll
    for (int u = 0; u < 4; ++u) {
      const int gm0 = m0 + wm + t * 16 + hi * 8;
      const int gn  = n0 + wn + u * 16 + l16;
#pragma unroll
      for (int r = 0; r < 8; ++r) {
        const int gm = gm0 + r;
        float v = acc[t][u][r];
        if (MODE == 1) v += WKb[(size_t)(gn & 63) * 1024 + (gm & 1023)];
        if (MODE == 2) outF[(size_t)gm * 512 + gn] = v;
        else           outB[(size_t)gm * 512 + gn] = f2bf(v);
      }
    }
}

// ---------------------------------------------------------------------------
// Causal flash attention, DH=64, bf16 QKV, f32 accumulate.
// Block = 64 queries (2 waves x 32), key tiles of 64 streamed through LDS.
// grid = (S/64, B*H)
// ---------------------------------------------------------------------------
#define KS_STR 72   // padded LDS row stride (144B, 16B-aligned)

__global__ __launch_bounds__(64)
void flash_attn(const u16* __restrict__ Q, const u16* __restrict__ K,
                const u16* __restrict__ V, u16* __restrict__ H) {
  __shared__ u16 Ks[64 * KS_STR];          // [key][d]
  __shared__ u16 Vt[64 * KS_STR];          // [d][key] (transposed)
  __shared__ u16 Pw[2 * 32 * KS_STR];      // per-wave P scratch [q][key]

  const int tid  = threadIdx.x;
  const int lane = tid & 31;
  const int wave = tid >> 5;
  const int l16  = lane & 15;
  const int hi   = lane >> 4;
  const int bh = blockIdx.y;
  const int b  = bh >> 3;
  const int h  = bh & 7;
  const int q0 = blockIdx.x * 64;
  const int qw = q0 + wave * 32;
  const size_t headOff = (size_t)b * (1024 * 512) + (size_t)h * 64;

  // persistent Q A-fragments: 2 m-tiles x 2 k-chunks (d = c*32..c*32+31)
  Frag qf[2][2];
#pragma unroll
  for (int t = 0; t < 2; ++t)
#pragma unroll
    for (int c = 0; c < 2; ++c) {
      const u16* p = Q + headOff + (size_t)(qw + t * 16 + l16) * 512 + c * 32;
      qf[t][c].f[0] = *(const F4*)(p + hi * 16);
      qf[t][c].f[1] = *(const F4*)(p + hi * 16 + 8);
    }

  v8f o[2][4];
  float Mr[2][8], Lr[2][8];
#pragma unroll
  for (int t = 0; t < 2; ++t) {
#pragma unroll
    for (int jd = 0; jd < 4; ++jd) o[t][jd] = vzero();
#pragma unroll
    for (int r = 0; r < 8; ++r) { Mr[t][r] = -1e30f; Lr[t][r] = 0.f; }
  }

  const int nkt = q0 / 64 + 1;             // causal: key tiles 0..q0/64
  for (int kt = 0; kt < nkt; ++kt) {
    const int kbase = kt * 64;
    {
      // thread tid loads key row kbase+tid (64 bf16 = 8x16B)
      const F4* gk = (const F4*)(K + headOff + (size_t)(kbase + tid) * 512);
      F4* sk = (F4*)(Ks + tid * KS_STR);
#pragma unroll
      for (int i = 0; i < 8; ++i) sk[i] = gk[i];
      // V transposed into Vt[d][key]
      const F4* gv = (const F4*)(V + headOff + (size_t)(kbase + tid) * 512);
#pragma unroll
      for (int i = 0; i < 8; ++i) {
        F4 vv = gv[i];
        const u16* pv = (const u16*)&vv;
#pragma unroll
        for (int e = 0; e < 8; ++e)
          Vt[(i * 8 + e) * KS_STR + tid] = pv[e];
      }
    }
    // warm next K/V tile in cache while this tile's WMMAs run
    if (kt + 1 < nkt) {
      __builtin_prefetch(K + headOff + (size_t)(kbase + 64 + tid) * 512, 0, 0);
      __builtin_prefetch(V + headOff + (size_t)(kbase + 64 + tid) * 512, 0, 0);
    }
    __syncthreads();

    // scores S = Q * K^T : B-frag lane n=key, k=d contiguous from Ks rows
    v8f s[2][4];
#pragma unroll
    for (int j = 0; j < 4; ++j) {
      Frag kb0, kb1;
      const u16* kr = Ks + (j * 16 + l16) * KS_STR;
      kb0.f[0] = *(const F4*)(kr + hi * 16);
      kb0.f[1] = *(const F4*)(kr + hi * 16 + 8);
      kb1.f[0] = *(const F4*)(kr + 32 + hi * 16);
      kb1.f[1] = *(const F4*)(kr + 32 + hi * 16 + 8);
#pragma unroll
      for (int t = 0; t < 2; ++t) {
        v8f z = vzero();
        z = wmma_bf16(qf[t][0], kb0, z);
        z = wmma_bf16(qf[t][1], kb1, z);
        s[t][j] = z;
      }
    }

    // causal mask + 1/sqrt(64) scale (bias already folded into K)
#pragma unroll
    for (int t = 0; t < 2; ++t)
#pragma unroll
      for (int j = 0; j < 4; ++j)
#pragma unroll
        for (int r = 0; r < 8; ++r) {
          const int qq = qw + t * 16 + hi * 8 + r;
          const int kk = kbase + j * 16 + l16;
          const float v = s[t][j][r];
          s[t][j][r] = (kk <= qq) ? v * 0.125f : -1e30f;
        }

    // online softmax (row reductions across the 16-lane half)
#pragma unroll
    for (int t = 0; t < 2; ++t)
#pragma unroll
      for (int r = 0; r < 8; ++r) {
        float mx = s[t][0][r];
#pragma unroll
        for (int j = 1; j < 4; ++j) mx = fmaxf(mx, s[t][j][r]);
        mx = fmaxf(mx, __shfl_xor(mx, 1));
        mx = fmaxf(mx, __shfl_xor(mx, 2));
        mx = fmaxf(mx, __shfl_xor(mx, 4));
        mx = fmaxf(mx, __shfl_xor(mx, 8));
        const float newM  = fmaxf(Mr[t][r], mx);
        const float alpha = __expf(Mr[t][r] - newM);
        float rs = 0.f;
#pragma unroll
        for (int j = 0; j < 4; ++j) {
          const float p = __expf(s[t][j][r] - newM);
          s[t][j][r] = p;
          rs += p;
        }
        rs += __shfl_xor(rs, 1);
        rs += __shfl_xor(rs, 2);
        rs += __shfl_xor(rs, 4);
        rs += __shfl_xor(rs, 8);
        Lr[t][r] = Lr[t][r] * alpha + rs;
        Mr[t][r] = newM;
#pragma unroll
        for (int jd = 0; jd < 4; ++jd) o[t][jd][r] *= alpha;
      }

    // P: C-fragment layout -> A-operand layout via per-wave LDS round trip
    u16* pb = Pw + wave * (32 * KS_STR);
#pragma unroll
    for (int t = 0; t < 2; ++t)
#pragma unroll
      for (int j = 0; j < 4; ++j)
#pragma unroll
        for (int r = 0; r < 8; ++r)
          pb[(t * 16 + hi * 8 + r) * KS_STR + j * 16 + l16] = f2bf(s[t][j][r]);

    asm volatile("s_wait_dscnt 0x0" ::: "memory");  // same-wave LDS RAW

    Frag pa[2][2];
#pragma unroll
    for (int t = 0; t < 2; ++t)
#pragma unroll
      for (int c = 0; c < 2; ++c) {
        const u16* p = pb + (t * 16 + l16) * KS_STR + c * 32;
        pa[t][c].f[0] = *(const F4*)(p + hi * 16);
        pa[t][c].f[1] = *(const F4*)(p + hi * 16 + 8);
      }

    // O += P * V : B-frag lane n=d, k=key contiguous from Vt rows
#pragma unroll
    for (int jd = 0; jd < 4; ++jd) {
      Frag vb0, vb1;
      const u16* vr = Vt + (jd * 16 + l16) * KS_STR;
      vb0.f[0] = *(const F4*)(vr + hi * 16);
      vb0.f[1] = *(const F4*)(vr + hi * 16 + 8);
      vb1.f[0] = *(const F4*)(vr + 32 + hi * 16);
      vb1.f[1] = *(const F4*)(vr + 32 + hi * 16 + 8);
#pragma unroll
      for (int t = 0; t < 2; ++t) {
        o[t][jd] = wmma_bf16(pa[t][0], vb0, o[t][jd]);
        o[t][jd] = wmma_bf16(pa[t][1], vb1, o[t][jd]);
      }
    }
    __syncthreads();
  }

  // normalize and store H (bf16, merged-head layout [b*S+s][h*64+d])
#pragma unroll
  for (int t = 0; t < 2; ++t)
#pragma unroll
    for (int jd = 0; jd < 4; ++jd)
#pragma unroll
      for (int r = 0; r < 8; ++r) {
        const int row = qw + t * 16 + hi * 8 + r;
        const int col = jd * 16 + l16;
        const float val = o[t][jd][r] / Lr[t][r];
        H[headOff + (size_t)row * 512 + col] = f2bf(val);
      }
}

// ---------------------------------------------------------------------------
extern "C" void kernel_launch(void* const* d_in, const int* in_sizes, int n_in,
                              void* d_out, int out_size, void* d_ws, size_t ws_size,
                              hipStream_t stream) {
  (void)in_sizes; (void)n_in; (void)out_size; (void)ws_size;
  const float* xs  = (const float*)d_in[0];
  const float* xt  = (const float*)d_in[1];
  const float* Wqs = (const float*)d_in[2];
  const float* Wks = (const float*)d_in[3];
  const float* Wvs = (const float*)d_in[4];
  const float* Wqt = (const float*)d_in[5];
  const float* Wkt = (const float*)d_in[6];
  const float* Wvt = (const float*)d_in[7];
  const float* WK  = (const float*)d_in[8];
  const float* Wo  = (const float*)d_in[9];
  float* out = (float*)d_out;

  char* ws = (char*)d_ws;
  size_t off = 0;
  u16* Xcat = (u16*)(ws + off); off += (size_t)8192 * 1024 * 2;  // 16 MB
  u16* WqT  = (u16*)(ws + off); off += (size_t)512 * 1024 * 2;
  u16* WkT  = (u16*)(ws + off); off += (size_t)512 * 1024 * 2;
  u16* WvT  = (u16*)(ws + off); off += (size_t)512 * 1024 * 2;
  u16* WoT  = (u16*)(ws + off); off += (size_t)512 * 512 * 2;
  u16* Qb   = (u16*)(ws + off); off += (size_t)8192 * 512 * 2;
  u16* Kb   = (u16*)(ws + off); off += (size_t)8192 * 512 * 2;
  u16* Vb   = (u16*)(ws + off); off += (size_t)8192 * 512 * 2;
  u16* Hb   = (u16*)(ws + off); off += (size_t)8192 * 512 * 2;   // ~54 MB total

  convert_x<<<32768, 256, 0, stream>>>(xs, xt, Xcat);
  convert_w<<<2048, 256, 0, stream>>>(Wqs, Wqt, WqT, 1024);
  convert_w<<<2048, 256, 0, stream>>>(Wks, Wkt, WkT, 1024);
  convert_w<<<2048, 256, 0, stream>>>(Wvs, Wvt, WvT, 1024);
  convert_w<<<1024, 256, 0, stream>>>(Wo,  Wo,  WoT, 512);

  dim3 g(64, 4);  // M/128, N/128
  gemm_bf16<0><<<g, 128, 0, stream>>>(Xcat, WqT, Qb, nullptr, WK, 1024);
  gemm_bf16<1><<<g, 128, 0, stream>>>(Xcat, WkT, Kb, nullptr, WK, 1024); // +WK
  gemm_bf16<0><<<g, 128, 0, stream>>>(Xcat, WvT, Vb, nullptr, WK, 1024);

  flash_attn<<<dim3(16, 64), 64, 0, stream>>>(Qb, Kb, Vb, Hb);

  gemm_bf16<2><<<g, 128, 0, stream>>>(Hb, WoT, nullptr, out, WK, 512);
}
